// AttentionHead_18494129177025
// MI455X (gfx1250) — matrix operand order repaired and checked by reading, not compiled, not run on hardware
//
#include <hip/hip_runtime.h>
#include <hip/hip_bf16.h>
#include <cmath>

typedef __attribute__((ext_vector_type(16))) __bf16 v16bf;
typedef __attribute__((ext_vector_type(8)))  __bf16 v8bf;
typedef __attribute__((ext_vector_type(8)))  float  v8f;

#define HIDDEN 1024
#define OUT    128
#define BATCH  4
#define SEQ    2048
#define ROWS   (BATCH*SEQ)   /* 8192 */

static __device__ __forceinline__ v16bf cat8(v8bf lo, v8bf hi) {
  return __builtin_shufflevector(lo, hi, 0,1,2,3,4,5,6,7,8,9,10,11,12,13,14,15);
}

static __device__ __forceinline__ v8f wmma_bf16(v16bf a, v16bf b, v8f c) {
  // D = A(16x32) * B(32x16) + C, fp32 accumulate
  return __builtin_amdgcn_wmma_f32_16x16x32_bf16(false, a, false, b, (short)0, c, false, false);
}

// ---------------------------------------------------------------------------
// Projection: Y[r,n] = (sum_k X[r,k]*W[k,n] + bias[n]) * scale, Y stored bf16.
// One wave computes one 16x16 tile. X:[8192,1024] f32, W:[1024,128] f32.
// ---------------------------------------------------------------------------
__global__ __launch_bounds__(256) void proj_kernel(const float* __restrict__ X,
                                                   const float* __restrict__ W,
                                                   const float* __restrict__ bias,
                                                   __bf16* __restrict__ Y,
                                                   float scale)
{
  const int lane  = threadIdx.x & 31;
  const int wv    = threadIdx.x >> 5;
  const int w     = blockIdx.x * 8 + wv;          // 4096 tiles total
  const int tileM = w >> 3;
  const int tileN = w & 7;
  const int nl    = lane & 15;
  const int hi    = (lane & 16) ? 1 : 0;
  const int kloA  = hi * 16;                       // A-frag K offset per half-wave
  const int kloB  = hi * 16;                       // B-frag K offset per half-wave
  const int arow  = tileM * 16 + nl;

  v8f acc = {};
  for (int k0 = 0; k0 < HIDDEN; k0 += 32) {
    // A fragment (16x32) from X rows, fp32 -> bf16
    v16bf a;
    #pragma unroll
    for (int e = 0; e < 8; ++e) {
      a[e]     = (__bf16)X[arow * HIDDEN + k0 + kloA + e];
      a[e + 8] = (__bf16)X[arow * HIDDEN + k0 + kloA + 8 + e];
    }
    // B fragment (32x16) from W, column nl, 16 consecutive K per half-wave
    v16bf bf;
    #pragma unroll
    for (int e = 0; e < 16; ++e)
      bf[e] = (__bf16)W[(k0 + kloB + e) * OUT + tileN * 16 + nl];
    acc = wmma_bf16(a, bf, acc);
  }

  const float bb = bias[tileN * 16 + nl];
  #pragma unroll
  for (int r = 0; r < 8; ++r) {
    float val = (acc[r] + bb) * scale;
    Y[(tileM * 16 + r + hi * 8) * OUT + tileN * 16 + nl] = (__bf16)val;
  }
}

// ---------------------------------------------------------------------------
// Flash attention: one wave per 16-query tile, stream 32-key blocks.
// Q,K,V: bf16 [8192,128] (batch-major, scale folded into Q). Out: f32.
// ---------------------------------------------------------------------------
__global__ __launch_bounds__(256) void attn_kernel(const __bf16* __restrict__ Q,
                                                   const __bf16* __restrict__ K,
                                                   const __bf16* __restrict__ V,
                                                   float* __restrict__ Out)
{
  __shared__ __bf16 plds[8][16 * 32];              // per-wave P tile (C->A relayout)

  const int lane = threadIdx.x & 31;
  const int wv   = threadIdx.x >> 5;
  const int w    = blockIdx.x * 8 + wv;            // 512 waves
  const int b    = w >> 7;                         // batch
  const int qt   = w & 127;                        // query tile in sequence
  const int nl   = lane & 15;
  const int hi   = (lane & 16) ? 1 : 0;
  const int kloA = hi * 16;
  const int kloB = hi * 16;
  const int qrow = b * SEQ + qt * 16 + nl;         // A-layout row for this lane

  // Q fragments: 4 x (16x32 bf16), contiguous 8-element runs -> b128 loads
  v16bf qf[4];
  #pragma unroll
  for (int f = 0; f < 4; ++f) {
    const v8bf* p0 = (const v8bf*)(Q + qrow * OUT + f * 32 + kloA);
    const v8bf* p1 = (const v8bf*)(Q + qrow * OUT + f * 32 + kloA + 8);
    qf[f] = cat8(p0[0], p1[0]);
  }

  v8f zero = {};
  v8f o[8];
  float m[8], l[8];
  #pragma unroll
  for (int t = 0; t < 8; ++t) o[t] = zero;
  #pragma unroll
  for (int r = 0; r < 8; ++r) { m[r] = -1e30f; l[r] = 0.0f; }

  for (int key0 = 0; key0 < SEQ; key0 += 32) {
    const int kb = b * SEQ + key0;

    // ---- scores: S(16x32) = Q(16x128) @ K_block^T(128x32) ----
    v8f s0 = zero, s1 = zero;
    #pragma unroll
    for (int f = 0; f < 4; ++f) {
      const __bf16* base0 = K + (kb + nl) * OUT + f * 32 + kloB;        // keys 0..15
      v16bf kb0 = cat8(((const v8bf*)base0)[0], ((const v8bf*)base0)[1]);
      s0 = wmma_bf16(qf[f], kb0, s0);
      const __bf16* base1 = K + (kb + 16 + nl) * OUT + f * 32 + kloB;   // keys 16..31
      v16bf kb1 = cat8(((const v8bf*)base1)[0], ((const v8bf*)base1)[1]);
      s1 = wmma_bf16(qf[f], kb1, s1);
    }

    // ---- online softmax (row reductions across the 16-lane half) ----
    float p0[8], p1[8], sc[8];
    #pragma unroll
    for (int r = 0; r < 8; ++r) {
      float t = fmaxf(s0[r], s1[r]);
      t = fmaxf(t, __shfl_xor(t, 1));
      t = fmaxf(t, __shfl_xor(t, 2));
      t = fmaxf(t, __shfl_xor(t, 4));
      t = fmaxf(t, __shfl_xor(t, 8));
      const float mn = fmaxf(m[r], t);
      sc[r] = __expf(m[r] - mn);
      m[r]  = mn;
      p0[r] = __expf(s0[r] - mn);
      p1[r] = __expf(s1[r] - mn);
      float rs = p0[r] + p1[r];
      rs += __shfl_xor(rs, 1);
      rs += __shfl_xor(rs, 2);
      rs += __shfl_xor(rs, 4);
      rs += __shfl_xor(rs, 8);
      l[r] = l[r] * sc[r] + rs;
    }
    #pragma unroll
    for (int t = 0; t < 8; ++t)
      #pragma unroll
      for (int r = 0; r < 8; ++r)
        o[t][r] *= sc[r];

    // ---- relayout P: C-layout regs -> LDS -> A-layout fragment (bf16) ----
    #pragma unroll
    for (int r = 0; r < 8; ++r) {
      const int prow = r + hi * 8;
      plds[wv][prow * 32 + nl]      = (__bf16)p0[r];
      plds[wv][prow * 32 + nl + 16] = (__bf16)p1[r];
    }
    __builtin_amdgcn_wave_barrier();
    asm volatile("s_wait_dscnt 0" ::: "memory");
    v16bf pa;
    #pragma unroll
    for (int e = 0; e < 8; ++e) {
      pa[e]     = plds[wv][nl * 32 + kloA + e];
      pa[e + 8] = plds[wv][nl * 32 + kloA + 8 + e];
    }
    __builtin_amdgcn_wave_barrier();

    // ---- O(16x128) += P(16x32) @ V_block(32x128) ----
    #pragma unroll
    for (int t = 0; t < 8; ++t) {
      v16bf vb;
      #pragma unroll
      for (int e = 0; e < 16; ++e)
        vb[e] = V[(kb + kloB + e) * OUT + t * 16 + nl];
      o[t] = wmma_bf16(pa, vb, o[t]);
    }
  }

  // ---- normalize and store fp32 output ----
  float linv[8];
  #pragma unroll
  for (int r = 0; r < 8; ++r) linv[r] = 1.0f / l[r];
  const int orow = b * SEQ + qt * 16;
  #pragma unroll
  for (int t = 0; t < 8; ++t)
    #pragma unroll
    for (int r = 0; r < 8; ++r)
      Out[(orow + r + hi * 8) * OUT + t * 16 + nl] = o[t][r] * linv[r];
}

extern "C" void kernel_launch(void* const* d_in, const int* in_sizes, int n_in,
                              void* d_out, int out_size, void* d_ws, size_t ws_size,
                              hipStream_t stream) {
  const float* q  = (const float*)d_in[0];
  const float* k  = (const float*)d_in[1];
  const float* v  = (const float*)d_in[2];
  const float* Wq = (const float*)d_in[3];
  const float* bq = (const float*)d_in[4];
  const float* Wk = (const float*)d_in[5];
  const float* bk = (const float*)d_in[6];
  const float* Wv = (const float*)d_in[7];
  const float* bv = (const float*)d_in[8];
  float* out = (float*)d_out;

  __bf16* Qb = (__bf16*)d_ws;                       // 8192x128 bf16 = 2 MB
  __bf16* Kb = Qb + (size_t)ROWS * OUT;
  __bf16* Vb = Kb + (size_t)ROWS * OUT;

  const float qscale = 1.0f / sqrtf((float)OUT);

  // 4096 tiles / 8 waves per block = 512 blocks per projection
  proj_kernel<<<512, 256, 0, stream>>>(q, Wq, bq, Qb, qscale);
  proj_kernel<<<512, 256, 0, stream>>>(k, Wk, bk, Kb, 1.0f);
  proj_kernel<<<512, 256, 0, stream>>>(v, Wv, bv, Vb, 1.0f);

  // 512 query-tile waves / 8 per block = 64 blocks
  attn_kernel<<<64, 256, 0, stream>>>(Qb, Kb, Vb, out);
}
